// CylinderFeat_29042568856084
// MI455X (gfx1250) — compile-verified, matrix operand order checked by reading, not run on hardware
//
#include <hip/hip_runtime.h>

typedef __attribute__((ext_vector_type(16))) _Float16 v16h;
typedef __attribute__((ext_vector_type(8)))  _Float16 v8h;
typedef __attribute__((ext_vector_type(8)))  float    v8f;

#define NPTS 500000
#define GX 480
#define GY 360
#define GZ 32
#define GRID_TOTAL (4*GX*GY*GZ)          /* 22,118,400 cells  */
#define NWORDS (GRID_TOTAL/32)           /* 691,200 bitmap words */
#define SCAN_BLK 256
#define NSCANBLK (NWORDS/SCAN_BLK)       /* 2700 */
#define ROWS 80                          /* rows per GEMM block (5 waves x 16) */
#define GEMM_THREADS 160
#define NGEMMBLK (NPTS/ROWS)             /* 6250, exact */

// ---------------- workspace layout (bytes) ----------------
#define OFF_Y3   ((size_t)0)                     // N*256 fp16 = 256 MB
#define OFF_Y2   ((size_t)256000000)             // N*128 fp16 = 128 MB
#define OFF_X4   OFF_Y2                          // alias: N*64 f32 (y2 dead by layer4)
#define OFF_Y1   ((size_t)384000000)             // N*64 fp16 = 64 MB
#define OFF_POOL ((size_t)448000000)             // N*64 u32 = 128 MB
#define OFF_LIN  ((size_t)576000000)             // N u32
#define OFF_INV  ((size_t)578000000)             // N u32
#define OFF_BM   ((size_t)580000000)             // bitmap
#define OFF_WPFX ((size_t)582764800)             // word prefix
#define OFF_BS   ((size_t)585529600)             // block sums
#define OFF_STATS ((size_t)585540400)            // sum/sq: 3,3,64,64,128,128,256,256
#define OFF_SS    ((size_t)585544008)            // scale/shift per stage
#define OFF_WH    ((size_t)585547616)            // fp16 weights: w2,w3,w4,wc (16B aligned)

// ---------------- CDNA5 async global->LDS copy (ASYNCcnt path) ----------------
#if defined(__gfx1250__) && __has_builtin(__builtin_amdgcn_global_load_async_to_lds_b128)
#define HAVE_ASYNC_LDS 1
typedef int v4i __attribute__((__vector_size__(4 * sizeof(int))));
typedef __attribute__((address_space(1))) v4i* g_v4i_p;   // global-AS vector ptr
typedef __attribute__((address_space(3))) v4i* l_v4i_p;   // LDS-AS vector ptr
__device__ __forceinline__ void async_copy16(const void* g, void* l) {
  __builtin_amdgcn_global_load_async_to_lds_b128(
      (g_v4i_p)(uintptr_t)g, (l_v4i_p)l, 0, 0);
}
__device__ __forceinline__ void async_wait0() {
#if __has_builtin(__builtin_amdgcn_s_wait_asynccnt)
  __builtin_amdgcn_s_wait_asynccnt(0);
#else
  asm volatile("s_wait_asynccnt 0x0" ::: "memory");
#endif
}
#endif

__device__ __forceinline__ v16h frag_ld(const _Float16* p16a, const _Float16* p16b) {
  v8h lo = *(const v8h*)p16a;
  v8h hi = *(const v8h*)p16b;
  return __builtin_shufflevector(lo, hi, 0, 1, 2, 3, 4, 5, 6, 7, 8, 9, 10, 11, 12, 13, 14, 15);
}

// ---------------- utility kernels ----------------
__global__ void k_zero_u32(unsigned* p, size_t n) {
  size_t i = (size_t)blockIdx.x * blockDim.x + threadIdx.x;
  size_t stride = (size_t)gridDim.x * blockDim.x;
  for (; i < n; i += stride) p[i] = 0u;
}
__global__ void k_fill_f32(float* p, size_t n, float v) {
  size_t i = (size_t)blockIdx.x * blockDim.x + threadIdx.x;
  size_t stride = (size_t)gridDim.x * blockDim.x;
  for (; i < n; i += stride) p[i] = v;
}
__global__ void k_cvt_h(const float* __restrict__ s, _Float16* __restrict__ d, int n) {
  int i = blockIdx.x * blockDim.x + threadIdx.x;
  if (i < n) d[i] = (_Float16)s[i];
}

// ---------------- BN statistics ----------------
__global__ void k_stats0(const float* __restrict__ fea, float* sum0, float* sq0, int n) {
  __shared__ float s[3], q[3];
  if (threadIdx.x < 3) { s[threadIdx.x] = 0.f; q[threadIdx.x] = 0.f; }
  __syncthreads();
  int i = blockIdx.x * blockDim.x + threadIdx.x;
  if (i < n) {
    #pragma unroll
    for (int c = 0; c < 3; ++c) {
      float v = fea[(size_t)i * 3 + c];
      atomicAdd(&s[c], v); atomicAdd(&q[c], v * v);
    }
  }
  __syncthreads();
  if (threadIdx.x < 3) { atomicAdd(&sum0[threadIdx.x], s[threadIdx.x]); atomicAdd(&sq0[threadIdx.x], q[threadIdx.x]); }
}

__global__ void k_finalize(const float* __restrict__ sum, const float* __restrict__ sq,
                           const float* __restrict__ g, const float* __restrict__ b,
                           float* scale, float* shift, int d, float invN) {
  int c = blockIdx.x * blockDim.x + threadIdx.x;
  if (c < d) {
    float m = sum[c] * invN;
    float var = fmaxf(sq[c] * invN - m * m, 0.f);
    float s = g[c] * rsqrtf(var + 1e-5f);
    scale[c] = s;
    shift[c] = b[c] - m * s;
  }
}

// ---------------- layer 1: BN0 -> Linear(3->64), scalar (K=3 too small for WMMA) ----------------
__global__ void k_layer1(const float* __restrict__ fea,
                         const float* __restrict__ sc0, const float* __restrict__ sh0,
                         const float* __restrict__ w1, const float* __restrict__ b1,
                         _Float16* __restrict__ y1, float* sum1, float* sq1, int n) {
  __shared__ float sw[3 * 64];
  __shared__ float sb[64], ssum[64], ssq[64];
  int tid = threadIdx.x;
  for (int i = tid; i < 192; i += blockDim.x) sw[i] = w1[i];
  for (int i = tid; i < 64; i += blockDim.x) { sb[i] = b1[i]; ssum[i] = 0.f; ssq[i] = 0.f; }
  __syncthreads();
  int i = blockIdx.x * blockDim.x + tid;
  if (i < n) {
    float x0 = fea[(size_t)i * 3 + 0] * sc0[0] + sh0[0];
    float x1 = fea[(size_t)i * 3 + 1] * sc0[1] + sh0[1];
    float x2 = fea[(size_t)i * 3 + 2] * sc0[2] + sh0[2];
    #pragma unroll
    for (int j = 0; j < 8; ++j) {
      v8h o;
      #pragma unroll
      for (int t = 0; t < 8; ++t) {
        int c = j * 8 + t;
        float v = fmaf(x0, sw[c], fmaf(x1, sw[64 + c], fmaf(x2, sw[128 + c], sb[c])));
        o[t] = (_Float16)v;
        atomicAdd(&ssum[c], v);
        atomicAdd(&ssq[c], v * v);
      }
      *(v8h*)&y1[(size_t)i * 64 + j * 8] = o;
    }
  }
  __syncthreads();
  for (int c = tid; c < 64; c += blockDim.x) { atomicAdd(&sum1[c], ssum[c]); atomicAdd(&sq1[c], ssq[c]); }
}

// ---------------- WMMA GEMM layers: A=relu(bn(yin)) [N,K] x W [K,NOUT] + b ----------------
template <int K, int NOUT, bool LAST>
__global__ void k_mlp_gemm(const _Float16* __restrict__ yin,
                           const float* __restrict__ scale, const float* __restrict__ shift,
                           const _Float16* __restrict__ wh,   // [K][NOUT]
                           const float* __restrict__ bias,    // [NOUT]
                           _Float16* __restrict__ yout_h,     // !LAST
                           float* __restrict__ yout_f,        // LAST
                           float* __restrict__ osum, float* __restrict__ osq) {
  __shared__ __align__(16) _Float16 sA[ROWS * K];
  __shared__ __align__(16) _Float16 sBt[16 * K];  // transposed slice: [n][k]
  __shared__ float sScale[K], sShift[K];
  const int tid = threadIdx.x;
  const size_t rowBase = (size_t)blockIdx.x * ROWS;

#ifdef HAVE_ASYNC_LDS
  // kick off raw A-tile copy (ASYNCcnt, no VGPR data), overlap with scale staging
  for (int i8 = tid; i8 < ROWS * K / 8; i8 += blockDim.x)
    async_copy16(&yin[rowBase * K + i8 * 8], &sA[i8 * 8]);
  for (int c = tid; c < K; c += blockDim.x) { sScale[c] = scale[c]; sShift[c] = shift[c]; }
  async_wait0();
  __syncthreads();
  // in-place BN affine + ReLU (fp16 -> fp16, same footprint)
  for (int i8 = tid; i8 < ROWS * K / 8; i8 += blockDim.x) {
    const int base = i8 * 8;
    const int c = base % K;
    v8h h = *(const v8h*)&sA[base];
    v8h o;
    #pragma unroll
    for (int j = 0; j < 8; ++j) {
      float v = fmaf((float)h[j], sScale[c + j], sShift[c + j]);
      o[j] = (_Float16)(v > 0.f ? v : 0.f);
    }
    *(v8h*)&sA[base] = o;
  }
#else
  for (int c = tid; c < K; c += blockDim.x) { sScale[c] = scale[c]; sShift[c] = shift[c]; }
  __syncthreads();
  for (int i8 = tid; i8 < ROWS * K / 8; i8 += blockDim.x) {
    const int base = i8 * 8;
    const int c = base % K;
    v8h h = *(const v8h*)&yin[rowBase * K + base];
    v8h o;
    #pragma unroll
    for (int j = 0; j < 8; ++j) {
      float v = fmaf((float)h[j], sScale[c + j], sShift[c + j]);
      o[j] = (_Float16)(v > 0.f ? v : 0.f);
    }
    *(v8h*)&sA[base] = o;
  }
#endif

  const int wave = tid >> 5, lane = tid & 31;
  const int l = lane & 15, hi = lane >> 4;
  const int wrow = wave * 16;
  for (int ct = 0; ct < NOUT / 16; ++ct) {
    __syncthreads();
    // stage B column-slice transposed so fragment reads are contiguous b128
    for (int i = tid; i < K * 16; i += blockDim.x) {
      int nn = i / K, k = i % K;
      sBt[i] = wh[(size_t)k * NOUT + ct * 16 + nn];
    }
    if (ct + 1 < NOUT / 16)
      __builtin_prefetch(&wh[(size_t)(tid % K) * NOUT + (ct + 1) * 16], 0, 1);
    __syncthreads();
    v8f acc = {};
    #pragma unroll
    for (int kc = 0; kc < K / 32; ++kc) {
      const int k0 = kc * 32;
      v16h a = frag_ld(&sA[(wrow + l) * K + k0 + hi * 8],
                       &sA[(wrow + l) * K + k0 + 16 + hi * 8]);
      v16h b = frag_ld(&sBt[l * K + k0 + hi * 8],
                       &sBt[l * K + k0 + 16 + hi * 8]);
      acc = __builtin_amdgcn_wmma_f32_16x16x32_f16(false, a, false, b, (short)0, acc, false, false);
    }
    const int col = ct * 16 + l;
    const float bcol = bias[col];
    float s = 0.f, s2 = 0.f;
    #pragma unroll
    for (int r = 0; r < 8; ++r) {
      float v = acc[r] + bcol;
      size_t grow = rowBase + wrow + r + 8 * hi;
      if constexpr (LAST) yout_f[grow * NOUT + col] = v;
      else                yout_h[grow * NOUT + col] = (_Float16)v;
      s += v; s2 += v * v;
    }
    if constexpr (!LAST) { atomicAdd(&osum[col], s); atomicAdd(&osq[col], s2); }
  }
}

// ---------------- voxel id + occupancy bitmap ----------------
__global__ void k_lin(const int* __restrict__ ind, unsigned* __restrict__ lin,
                      unsigned* __restrict__ bm, int n) {
  int i = blockIdx.x * blockDim.x + threadIdx.x;
  if (i >= n) return;
  int x = ind[(size_t)i * 4 + 0], y = ind[(size_t)i * 4 + 1];
  int z = ind[(size_t)i * 4 + 2], b = ind[(size_t)i * 4 + 3];
  unsigned L = (((unsigned)(b * GX + x) * GY + y) * GZ) + (unsigned)z;
  lin[i] = L;
  atomicOr(&bm[L >> 5], 1u << (L & 31));
}

__global__ void k_scan1(const unsigned* __restrict__ bm, unsigned* __restrict__ bs) {
  __shared__ unsigned s[SCAN_BLK];
  int w = blockIdx.x * SCAN_BLK + threadIdx.x;
  s[threadIdx.x] = (w < NWORDS) ? (unsigned)__popc(bm[w]) : 0u;
  __syncthreads();
  for (int off = SCAN_BLK / 2; off > 0; off >>= 1) {
    if (threadIdx.x < off) s[threadIdx.x] += s[threadIdx.x + off];
    __syncthreads();
  }
  if (threadIdx.x == 0) bs[blockIdx.x] = s[0];
}

__global__ void k_scan2(unsigned* bs, int nblk) {
  if (blockIdx.x == 0 && threadIdx.x == 0) {
    unsigned acc = 0;
    for (int i = 0; i < nblk; ++i) { unsigned t = bs[i]; bs[i] = acc; acc += t; }
  }
}

// word prefix + emit sorted-unique rows (decoded) into d_out
__global__ void k_scan3(const unsigned* __restrict__ bm, const unsigned* __restrict__ bs,
                        unsigned* __restrict__ wpfx, float* __restrict__ unq_out) {
  __shared__ unsigned s[SCAN_BLK];
  int w = blockIdx.x * SCAN_BLK + threadIdx.x;
  unsigned word = (w < NWORDS) ? bm[w] : 0u;
  unsigned c = (unsigned)__popc(word);
  s[threadIdx.x] = c;
  __syncthreads();
  for (int off = 1; off < SCAN_BLK; off <<= 1) {
    unsigned t = (threadIdx.x >= (unsigned)off) ? s[threadIdx.x - off] : 0u;
    __syncthreads();
    s[threadIdx.x] += t;
    __syncthreads();
  }
  if (w >= NWORDS) return;
  unsigned excl = s[threadIdx.x] - c + bs[blockIdx.x];
  wpfx[w] = excl;
  unsigned rem = word, rank = excl;
  while (rem) {
    int bit = __builtin_ctz(rem);
    rem &= rem - 1u;
    unsigned L = (unsigned)w * 32u + (unsigned)bit;
    unsigned b_ = L / (GX * GY * GZ); unsigned r = L % (GX * GY * GZ);
    unsigned x_ = r / (GY * GZ); r %= (GY * GZ);
    unsigned y_ = r / GZ, z_ = r % GZ;
    unq_out[(size_t)rank * 4 + 0] = (float)b_;
    unq_out[(size_t)rank * 4 + 1] = (float)x_;
    unq_out[(size_t)rank * 4 + 2] = (float)y_;
    unq_out[(size_t)rank * 4 + 3] = (float)z_;
    ++rank;
  }
}

__global__ void k_inv(const unsigned* __restrict__ lin, const unsigned* __restrict__ bm,
                      const unsigned* __restrict__ wpfx, unsigned* __restrict__ inv_u,
                      float* __restrict__ inv_f, int n) {
  int i = blockIdx.x * blockDim.x + threadIdx.x;
  if (i >= n) return;
  unsigned L = lin[i], w = L >> 5, bit = L & 31;
  unsigned rank = wpfx[w] + (unsigned)__popc(bm[w] & ((1u << bit) - 1u));
  inv_u[i] = rank;
  inv_f[i] = (float)rank;
}

// ---------------- segment max pooling (monotone u32 transform + atomicMax) ----------------
__global__ void k_pool(const float* __restrict__ x4, const unsigned* __restrict__ inv,
                       unsigned* __restrict__ pooled, int n) {
  size_t idx = (size_t)blockIdx.x * blockDim.x + threadIdx.x;
  if (idx >= (size_t)n * 64) return;
  int i = (int)(idx >> 6), f = (int)(idx & 63);
  unsigned u = __float_as_uint(x4[idx]);
  unsigned ord = (u & 0x80000000u) ? ~u : (u | 0x80000000u);
  atomicMax(&pooled[(size_t)inv[i] * 64 + f], ord);
}

// ---------------- compression: relu(pooled @ wc + bc) via WMMA ----------------
__global__ void k_compress(const unsigned* __restrict__ pooled,
                           const _Float16* __restrict__ wch, const float* __restrict__ bc,
                           float* __restrict__ out) {
  __shared__ __align__(16) _Float16 sA[ROWS * 64];
  __shared__ __align__(16) _Float16 sBt[16 * 64];  // transposed: [n][k]
#ifdef HAVE_ASYNC_LDS
  __shared__ __align__(16) unsigned sRaw[ROWS * 64];  // raw ord-u32 staging
#endif
  const int tid = threadIdx.x;
  const size_t rowBase = (size_t)blockIdx.x * ROWS;
#ifdef HAVE_ASYNC_LDS
  for (int i4 = tid; i4 < ROWS * 64 / 4; i4 += blockDim.x)
    async_copy16(&pooled[rowBase * 64 + i4 * 4], &sRaw[i4 * 4]);
  for (int i = tid; i < 64 * 16; i += blockDim.x) {
    int nn = i / 64, k = i % 64;
    sBt[i] = wch[(size_t)k * 16 + nn];
  }
  async_wait0();
  __syncthreads();
  for (int i = tid; i < ROWS * 64; i += blockDim.x) {
    unsigned o = sRaw[i];
    float f;
    if (o == 0u) f = 0.f;
    else if (o & 0x80000000u) f = __uint_as_float(o ^ 0x80000000u);
    else f = __uint_as_float(~o);
    sA[i] = (_Float16)f;
  }
#else
  for (int i = tid; i < 64 * 16; i += blockDim.x) {
    int nn = i / 64, k = i % 64;
    sBt[i] = wch[(size_t)k * 16 + nn];
  }
  for (int i = tid; i < ROWS * 64; i += blockDim.x) {
    unsigned o = pooled[rowBase * 64 + i];
    float f;
    if (o == 0u) f = 0.f;
    else if (o & 0x80000000u) f = __uint_as_float(o ^ 0x80000000u);
    else f = __uint_as_float(~o);
    sA[i] = (_Float16)f;
  }
#endif
  __syncthreads();
  const int wave = tid >> 5, lane = tid & 31;
  const int l = lane & 15, hi = lane >> 4;
  const int wrow = wave * 16;
  v8f acc = {};
  #pragma unroll
  for (int kc = 0; kc < 2; ++kc) {
    const int k0 = kc * 32;
    v16h a = frag_ld(&sA[(wrow + l) * 64 + k0 + hi * 8],
                     &sA[(wrow + l) * 64 + k0 + 16 + hi * 8]);
    v16h b = frag_ld(&sBt[l * 64 + k0 + hi * 8],
                     &sBt[l * 64 + k0 + 16 + hi * 8]);
    acc = __builtin_amdgcn_wmma_f32_16x16x32_f16(false, a, false, b, (short)0, acc, false, false);
  }
  const float bcol = bc[l];
  #pragma unroll
  for (int r = 0; r < 8; ++r) {
    float v = acc[r] + bcol;
    out[(rowBase + wrow + r + 8 * hi) * 16 + l] = v > 0.f ? v : 0.f;
  }
}

// ---------------- host orchestration ----------------
extern "C" void kernel_launch(void* const* d_in, const int* in_sizes, int n_in,
                              void* d_out, int out_size, void* d_ws, size_t ws_size,
                              hipStream_t stream) {
  (void)in_sizes; (void)n_in; (void)out_size; (void)ws_size;
  const float* fea = (const float*)d_in[0];
  const int*   ind = (const int*)d_in[1];
  const float* w1 = (const float*)d_in[2];  const float* b1 = (const float*)d_in[3];
  const float* w2 = (const float*)d_in[4];  const float* b2 = (const float*)d_in[5];
  const float* w3 = (const float*)d_in[6];  const float* b3 = (const float*)d_in[7];
  const float* w4 = (const float*)d_in[8];  const float* b4 = (const float*)d_in[9];
  const float* g0 = (const float*)d_in[10]; const float* bb0 = (const float*)d_in[11];
  const float* g1 = (const float*)d_in[12]; const float* bb1 = (const float*)d_in[13];
  const float* g2 = (const float*)d_in[14]; const float* bb2 = (const float*)d_in[15];
  const float* g3 = (const float*)d_in[16]; const float* bb3 = (const float*)d_in[17];
  const float* wc = (const float*)d_in[18]; const float* bc = (const float*)d_in[19];

  char* ws = (char*)d_ws;
  _Float16* y1 = (_Float16*)(ws + OFF_Y1);
  _Float16* y2 = (_Float16*)(ws + OFF_Y2);
  _Float16* y3 = (_Float16*)(ws + OFF_Y3);
  float*    x4 = (float*)(ws + OFF_X4);
  unsigned* pooled = (unsigned*)(ws + OFF_POOL);
  unsigned* lin = (unsigned*)(ws + OFF_LIN);
  unsigned* inv = (unsigned*)(ws + OFF_INV);
  unsigned* bm  = (unsigned*)(ws + OFF_BM);
  unsigned* wpfx = (unsigned*)(ws + OFF_WPFX);
  unsigned* bs  = (unsigned*)(ws + OFF_BS);
  float* stats = (float*)(ws + OFF_STATS);
  float* sum0 = stats;        float* sq0 = stats + 3;
  float* sum1 = stats + 6;    float* sq1 = stats + 70;
  float* sum2 = stats + 134;  float* sq2 = stats + 262;
  float* sum3 = stats + 390;  float* sq3 = stats + 646;   // end 902
  float* ss = (float*)(ws + OFF_SS);
  float* sc0 = ss;        float* sh0 = ss + 3;
  float* sc1 = ss + 6;    float* sh1 = ss + 70;
  float* sc2 = ss + 134;  float* sh2 = ss + 262;
  float* sc3 = ss + 390;  float* sh3 = ss + 646;
  _Float16* w2h = (_Float16*)(ws + OFF_WH);
  _Float16* w3h = w2h + 64 * 128;
  _Float16* w4h = w3h + 128 * 256;
  _Float16* wch = w4h + 256 * 64;

  float* out_unq = (float*)d_out;                        // N*4
  float* out_inv = (float*)d_out + (size_t)NPTS * 4;     // N
  float* out_cmp = (float*)d_out + (size_t)NPTS * 5;     // N*16

  const int n = NPTS;
  const int nb256 = (n + 255) / 256;
  const float invN = 1.0f / (float)n;

  // --- init scratch that is accumulated into ---
  k_zero_u32<<<2048, 256, 0, stream>>>(bm, (size_t)NWORDS);
  k_zero_u32<<<4096, 256, 0, stream>>>(pooled, (size_t)n * 64);
  k_zero_u32<<<4, 256, 0, stream>>>((unsigned*)stats, 902);
  k_fill_f32<<<2048, 256, 0, stream>>>(out_unq, (size_t)n * 4, -1.0f);

  // --- fp16 weight conversion ---
  k_cvt_h<<<(64 * 128 + 255) / 256, 256, 0, stream>>>(w2, w2h, 64 * 128);
  k_cvt_h<<<(128 * 256 + 255) / 256, 256, 0, stream>>>(w3, w3h, 128 * 256);
  k_cvt_h<<<(256 * 64 + 255) / 256, 256, 0, stream>>>(w4, w4h, 256 * 64);
  k_cvt_h<<<(64 * 16 + 255) / 256, 256, 0, stream>>>(wc, wch, 64 * 16);

  // --- unique via bitmap + popcount scan ---
  k_lin<<<nb256, 256, 0, stream>>>(ind, lin, bm, n);
  k_scan1<<<NSCANBLK, SCAN_BLK, 0, stream>>>(bm, bs);
  k_scan2<<<1, 32, 0, stream>>>(bs, NSCANBLK);
  k_scan3<<<NSCANBLK, SCAN_BLK, 0, stream>>>(bm, bs, wpfx, out_unq);
  k_inv<<<nb256, 256, 0, stream>>>(lin, bm, wpfx, inv, out_inv, n);

  // --- MLP with fused BN ---
  k_stats0<<<nb256, 256, 0, stream>>>(fea, sum0, sq0, n);
  k_finalize<<<1, 32, 0, stream>>>(sum0, sq0, g0, bb0, sc0, sh0, 3, invN);
  k_layer1<<<nb256, 256, 0, stream>>>(fea, sc0, sh0, w1, b1, y1, sum1, sq1, n);
  k_finalize<<<1, 64, 0, stream>>>(sum1, sq1, g1, bb1, sc1, sh1, 64, invN);
  k_mlp_gemm<64, 128, false><<<NGEMMBLK, GEMM_THREADS, 0, stream>>>(
      y1, sc1, sh1, w2h, b2, y2, nullptr, sum2, sq2);
  k_finalize<<<1, 128, 0, stream>>>(sum2, sq2, g2, bb2, sc2, sh2, 128, invN);
  k_mlp_gemm<128, 256, false><<<NGEMMBLK, GEMM_THREADS, 0, stream>>>(
      y2, sc2, sh2, w3h, b3, y3, nullptr, sum3, sq3);
  k_finalize<<<1, 256, 0, stream>>>(sum3, sq3, g3, bb3, sc3, sh3, 256, invN);
  k_mlp_gemm<256, 64, true><<<NGEMMBLK, GEMM_THREADS, 0, stream>>>(
      y3, sc3, sh3, w4h, b4, nullptr, x4, nullptr, nullptr);

  // --- segment max + compression ---
  k_pool<<<(int)(((size_t)n * 64 + 255) / 256), 256, 0, stream>>>(x4, inv, pooled, n);
  k_compress<<<NGEMMBLK, GEMM_THREADS, 0, stream>>>(pooled, wch, bc, out_cmp);
}
